// TransformerLinearApprox_88802743812165
// MI455X (gfx1250) — matrix operand order repaired and checked by reading, not compiled, not run on hardware
//
#include <hip/hip_runtime.h>
#include <math.h>

// Set to 0 if the tensor_load_to_lds inline asm is rejected by this toolchain.
#ifndef USE_TDM
#define USE_TDM 1
#endif

// ---------------------------------------------------------------------------
// Types for CDNA5 WMMA (wave32)
// ---------------------------------------------------------------------------
typedef __attribute__((ext_vector_type(16))) __bf16 v16bf;
typedef __attribute__((ext_vector_type(8)))  float  v8f;
typedef unsigned int u32x4 __attribute__((ext_vector_type(4)));
typedef int          i32x8 __attribute__((ext_vector_type(8)));

#define B_SZ 16
#define Dm   320
#define Nn   2048
#define EMBC 128
#define HSTRIDE ((long)Dm * Nn)   // 655360 elements per batch
#define WSTRIDE ((long)Dm * Dm)   // 102400 elements per layer weight
#define LDSP 48                   // padded LDS row stride (elements, 96B)

enum { F_ACCUM = 1, F_ZLAST = 2, F_TRANSC = 4 };

// ---------------------------------------------------------------------------
// Fragment loaders (contiguous k order, wave32)
// A: 16x32 bf16 (MxK). lane L<16: M=L, K=k0..k0+15
//                      lane L>=16: M=L-16, K=k0+16..k0+31.
// ---------------------------------------------------------------------------
__device__ __forceinline__ v16bf frag_a(const __bf16* __restrict__ A, int lda,
                                        int row_base, int k0, int lane) {
  int r    = row_base + (lane & 15);
  int koff = k0 + ((lane >> 4) << 4);
  const __bf16* p = A + (long)r * lda + koff;
  v16bf f;
  uint4* fp = reinterpret_cast<uint4*>(&f);
  fp[0] = *reinterpret_cast<const uint4*>(p);
  fp[1] = *reinterpret_cast<const uint4*>(p + 8);
  return f;
}

// B: 32x16 bf16 (KxN): lanes 0-15 hold K=0..15 (N=lane), lanes 16-31 K=16..31.
// Source: N-major tile in LDS, row n_local, LDSP-element stride.
__device__ __forceinline__ v16bf frag_b_lds(const __bf16* Bt, int s, int lane) {
  int n_local = s * 16 + (lane & 15);
  int kloc    = (lane >> 4) << 4;   // 0 or 16
  const __bf16* p = Bt + n_local * LDSP + kloc;
  v16bf f;
  uint4* fp = reinterpret_cast<uint4*>(&f);
  fp[0] = *reinterpret_cast<const uint4*>(p);
  fp[1] = *reinterpret_cast<const uint4*>(p + 8);
  return f;
}

// B = srcT: B[k,n] = src[n*ld + k]  (per-lane contiguous from global)
__device__ __forceinline__ v16bf frag_bT(const __bf16* __restrict__ src, int ld,
                                         int col_base, int k0, int lane) {
  int j  = col_base + (lane & 15);
  int kb = k0 + ((lane >> 4) << 4);
  const __bf16* p = src + (long)j * ld + kb;
  v16bf f;
  uint4* fp = reinterpret_cast<uint4*>(&f);
  fp[0] = *reinterpret_cast<const uint4*>(p);
  fp[1] = *reinterpret_cast<const uint4*>(p + 8);
  return f;
}

#define WMMA_BF16(acc, a, bfrag) \
  __builtin_amdgcn_wmma_f32_16x16x32_bf16(false, (a), false, (bfrag), (short)0, (acc), false, false)

__device__ __forceinline__ void wait_tensor0() {
#if __has_builtin(__builtin_amdgcn_s_wait_tensorcnt)
  __builtin_amdgcn_s_wait_tensorcnt(0);
#else
  asm volatile("s_wait_tensorcnt 0x0" ::: "memory");
#endif
}

#if USE_TDM
// Tensor Data Mover: copy a 64(n) x 32(k) bf16 tile (row stride ldbT elements)
// from global to LDS, inserting 16 pad elements after each 32-element row so
// the LDS layout is [64][LDSP=48].  D# per CDNA5 ISA.
__device__ __forceinline__ void tdm_stage_tile(const __bf16* src,
                                               unsigned lds_byte, int ldbT) {
  unsigned long long ga = (unsigned long long)(const void*)src;
  u32x4 g0;
  g0[0] = 1u;                                     // count=1, user desc
  g0[1] = lds_byte;                               // lds_addr (bytes)
  g0[2] = (unsigned)(ga & 0xFFFFFFFFull);         // global_addr[31:0]
  g0[3] = (unsigned)((ga >> 32) & 0x1FFFFFFull)   // global_addr[56:32]
        | (2u << 30);                             // type=2 (image)
  i32x8 g1;
  g1[0] = (int)((1u << 16)      // data_size = 1 -> 2 bytes
              | (1u << 20)      // pad_enable
              | (3u << 22)      // pad_interval: 16 DWORDs (=64B = one 32-elem row)
              | (7u << 25));    // pad_amount: 8 DWORDs (=16 elements)
  g1[1] = (int)(32u << 16);     // tensor_dim0 = 32 (bits 79:48, low half)
  g1[2] = (int)(64u << 16);     // tensor_dim0 hi=0 | tensor_dim1 = 64 (low half)
  g1[3] = (int)(32u << 16);     // tensor_dim1 hi=0 | tile_dim0 = 32
  g1[4] = 64;                   // tile_dim1 = 64, tile_dim2 = 0
  g1[5] = ldbT;                 // tensor_dim0_stride[31:0] (elements)
  g1[6] = 0;
  g1[7] = 0;
  asm volatile("tensor_load_to_lds %0, %1" :: "s"(g0), "s"(g1) : "memory");
}
#endif

// ---------------------------------------------------------------------------
// Generic GEMM: C(MxN) = alpha * A(MxK) @ B(KxN), bf16 in, f32 accumulate.
// B is supplied TRANSPOSED in memory (BT: N x K row-major) so each K-chunk is
// a rectangular 64x32 tile -> staged to LDS by the TDM (double-buffered).
// Block = 4 waves, 64x64 C tile; wave w owns rows [m0+16w,16) x 64 cols.
// Flags: F_ACCUM (C32 += ), F_ZLAST (zero bf16 col N-1), F_TRANSC (store C^T).
// ---------------------------------------------------------------------------
__global__ __launch_bounds__(128)
void gemm_bf16_lds(const __bf16* __restrict__ A, long strideA, int lda,
                   const __bf16* __restrict__ BT, long strideB, int ldbT,
                   float* __restrict__ C32, __bf16* __restrict__ Cb,
                   long strideC, int ldc,
                   int Nfull, int Kdim, float alpha, int flags) {
  const int b   = blockIdx.z;
  const int n0  = blockIdx.x * 64;
  const int m0  = blockIdx.y * 64;
  const __bf16* Ab = A  + b * strideA;
  const __bf16* Bb = BT + b * strideB;
  const int tid = threadIdx.x;
  const int lane = tid & 31, wave = tid >> 5;
  const int mw = m0 + wave * 16;

  __shared__ alignas(16) __bf16 Bt[2][64][LDSP];
  v8f acc0 = {}, acc1 = {}, acc2 = {}, acc3 = {};

  auto stage = [&](int buf, int k0) {
#if USE_TDM
    if (wave == 0) {
      tdm_stage_tile(Bb + (long)n0 * ldbT + k0,
                     (unsigned)(size_t)&Bt[buf][0][0], ldbT);
    }
#else
    int n = tid >> 1, c = tid & 1;
    const uint4* s4 = reinterpret_cast<const uint4*>(Bb + (long)(n0 + n) * ldbT + k0) + c;
    uint4* d4 = reinterpret_cast<uint4*>(&Bt[buf][n][0]) + c;
    d4[0] = s4[0];     // elements c*8   .. c*8+7
    d4[2] = s4[2];     // elements c*8+16.. c*8+23
#endif
  };

  stage(0, 0);
#if USE_TDM
  if (wave == 0) wait_tensor0();
#endif
  __syncthreads();

  int buf = 0;
  for (int k0 = 0; k0 < Kdim; k0 += 32) {
    if (k0 + 32 < Kdim) {
      stage(buf ^ 1, k0 + 32);
      __builtin_prefetch(Ab + (long)(mw + (lane & 15)) * lda + k0 + 32, 0, 1);
    }
    v16bf af = frag_a(Ab, lda, mw, k0, lane);
    const __bf16* base = &Bt[buf][0][0];
    v16bf b0 = frag_b_lds(base, 0, lane);
    acc0 = WMMA_BF16(acc0, af, b0);
    v16bf b1 = frag_b_lds(base, 1, lane);
    acc1 = WMMA_BF16(acc1, af, b1);
    v16bf b2 = frag_b_lds(base, 2, lane);
    acc2 = WMMA_BF16(acc2, af, b2);
    v16bf b3 = frag_b_lds(base, 3, lane);
    acc3 = WMMA_BF16(acc3, af, b3);
#if USE_TDM
    if (wave == 0) wait_tensor0();
#endif
    __syncthreads();
    buf ^= 1;
  }

  float*  Cp  = C32 ? C32 + b * strideC : nullptr;
  __bf16* Cbp = Cb  ? Cb  + b * strideC : nullptr;
  const int rb = mw + ((lane >> 4) << 3);   // C layout: VGPR v -> row v (+8 hi half)
  const int cb = n0 + (lane & 15);
  v8f accs[4] = {acc0, acc1, acc2, acc3};
  #pragma unroll
  for (int s = 0; s < 4; ++s) {
    #pragma unroll
    for (int v = 0; v < 8; ++v) {
      int row = rb + v, col = cb + s * 16;
      long ci = (flags & F_TRANSC) ? (long)col * ldc + row : (long)row * ldc + col;
      float val = accs[s][v] * alpha;
      if (flags & F_ACCUM) val += Cp[ci];
      if (Cp)  Cp[ci] = val;
      if (Cbp) {
        __bf16 o = (__bf16)val;
        if ((flags & F_ZLAST) && col == Nfull - 1) o = (__bf16)0.0f;
        Cbp[ci] = o;
      }
    }
  }
}

// ---------------------------------------------------------------------------
// S = alpha * V @ K^T (contraction over n; K column 2047 pre-zeroed).
// V, K are kept D x N row-major: both operands per-lane contiguous from global.
// ---------------------------------------------------------------------------
__global__ __launch_bounds__(128)
void gemm_vkt(const __bf16* __restrict__ V, const __bf16* __restrict__ K,
              __bf16* __restrict__ S, float alpha) {
  const int b  = blockIdx.z;
  const int n0 = blockIdx.x * 64;   // j
  const int m0 = blockIdx.y * 64;   // i
  const __bf16* Vb = V + (long)b * HSTRIDE;
  const __bf16* Kb = K + (long)b * HSTRIDE;
  const int tid = threadIdx.x, lane = tid & 31, wave = tid >> 5;
  const int mw = m0 + wave * 16;

  v8f acc0 = {}, acc1 = {}, acc2 = {}, acc3 = {};
  for (int k0 = 0; k0 < Nn; k0 += 32) {
    v16bf af = frag_a(Vb, Nn, mw, k0, lane);
    acc0 = WMMA_BF16(acc0, af, frag_bT(Kb, Nn, n0 +  0, k0, lane));
    acc1 = WMMA_BF16(acc1, af, frag_bT(Kb, Nn, n0 + 16, k0, lane));
    acc2 = WMMA_BF16(acc2, af, frag_bT(Kb, Nn, n0 + 32, k0, lane));
    acc3 = WMMA_BF16(acc3, af, frag_bT(Kb, Nn, n0 + 48, k0, lane));
  }
  __bf16* Sp = S + (long)b * WSTRIDE;
  const int rb = mw + ((lane >> 4) << 3);
  const int cb = n0 + (lane & 15);
  v8f accs[4] = {acc0, acc1, acc2, acc3};
  #pragma unroll
  for (int s = 0; s < 4; ++s)
    #pragma unroll
    for (int v = 0; v < 8; ++v)
      Sp[(long)(rb + v) * Dm + cb + s * 16] = (__bf16)(accs[s][v] * alpha);
}

// ---------------------------------------------------------------------------
// Prep / elementwise kernels
// ---------------------------------------------------------------------------
__global__ void cvt_f32_bf16(const float* __restrict__ s, __bf16* __restrict__ d, long n) {
  long i  = (long)blockIdx.x * blockDim.x + threadIdx.x;
  long st = (long)gridDim.x * blockDim.x;
  for (; i < n; i += st) d[i] = (__bf16)s[i];
}

__global__ void transpose_cvt128(const float* __restrict__ s, __bf16* __restrict__ dT) {
  int i = blockIdx.x * blockDim.x + threadIdx.x;   // 16384 threads exactly
  int c = i >> 7, d = i & 127;
  dT[i] = (__bf16)s[d * 128 + c];                  // dT[c][d] = Emb[d][c]
}

// xT[b][n][r] = bf16(x[b][64+r][n]),  r<128   (N-major activation layout)
__global__ void cvt_xmidT(const float* __restrict__ x, __bf16* __restrict__ xT) {
  const long n = (long)B_SZ * Nn * 128;
  long i  = (long)blockIdx.x * blockDim.x + threadIdx.x;
  long st = (long)gridDim.x * blockDim.x;
  for (; i < n; i += st) {
    long r = i & 127;
    long nn = (i >> 7) & (Nn - 1);
    long b = i >> 18;
    xT[i] = (__bf16)x[b * HSTRIDE + (64 + r) * Nn + nn];
  }
}

// hT cols 0..63 and 192..319 are identity-mapped from x (transposed store)
__global__ void copy_edgesT(const float* __restrict__ x, float* __restrict__ h32T,
                            __bf16* __restrict__ hbT) {
  const long n = (long)B_SZ * Nn * 192;
  long i  = (long)blockIdx.x * blockDim.x + threadIdx.x;
  long st = (long)gridDim.x * blockDim.x;
  for (; i < n; i += st) {
    long r = i % 192;
    long j = i / 192;
    long nn = j & (Nn - 1);
    long b = j >> 11;
    long row = (r < 64) ? r : (r + 128);
    float v = x[b * HSTRIDE + row * Nn + nn];
    long o = b * HSTRIDE + nn * Dm + row;
    h32T[o] = v;
    hbT[o]  = (__bf16)v;
  }
}

// logitsN is (B,2048,128) f32 row-contiguous; emit logits + softmax(logits).
__global__ __launch_bounds__(256)
void softmax_out(const float* __restrict__ lt, float* __restrict__ out) {
  long i = (long)blockIdx.x * blockDim.x + threadIdx.x;   // one (b,n) per thread
  if (i >= (long)B_SZ * Nn) return;
  const float* row = lt + i * EMBC;
  float* lo = out + i * EMBC;
  float* po = out + (long)B_SZ * Nn * EMBC + i * EMBC;
  float m = -3.402823466e38f;
  for (int c = 0; c < EMBC; ++c) {
    float v = row[c];
    lo[c] = v;
    m = fmaxf(m, v);
  }
  float ssum = 0.f;
  for (int c = 0; c < EMBC; ++c) ssum += __expf(row[c] - m);
  float inv = 1.f / ssum;
  for (int c = 0; c < EMBC; ++c) po[c] = __expf(row[c] - m) * inv;
}

// ---------------------------------------------------------------------------
// Host orchestration.  Activations (h, Q, attn) live transposed (B, N, D);
// K and V stay (B, D, N) for the V@K^T outer-product step.
// ---------------------------------------------------------------------------
extern "C" void kernel_launch(void* const* d_in, const int* in_sizes, int n_in,
                              void* d_out, int out_size, void* d_ws, size_t ws_size,
                              hipStream_t stream) {
  (void)in_sizes; (void)n_in; (void)out_size; (void)ws_size;
  const float* x   = (const float*)d_in[0];
  const float* emb = (const float*)d_in[1];
  const float* wq  = (const float*)d_in[2];
  const float* wk  = (const float*)d_in[3];
  const float* wv  = (const float*)d_in[4];
  const float* pp  = (const float*)d_in[5];
  float* out = (float*)d_out;

  char* ws = (char*)d_ws;
  float*  h32T = (float*)(ws);                         // (B,N,D) f32, 41,943,040 B
  __bf16* hbT  = (__bf16*)(ws +  41943040);            // (B,N,D) bf16
  __bf16* QbT  = (__bf16*)(ws +  62914560);            // (B,N,D) bf16 (xmidT first)
  __bf16* Kb   = (__bf16*)(ws +  83886080);            // (B,D,N) bf16 (attnT later)
  __bf16* Vb   = (__bf16*)(ws + 104857600);            // (B,D,N) bf16 (logitsN later)
  __bf16* Sb   = (__bf16*)(ws + 125829120);            // (B,D,D) bf16
  __bf16* Wqb  = (__bf16*)(ws + 129105920);
  __bf16* Wkb  = (__bf16*)(ws + 129720320);
  __bf16* Wvb  = (__bf16*)(ws + 130334720);
  __bf16* Pb   = (__bf16*)(ws + 130949120);
  __bf16* Embb  = (__bf16*)(ws + 131563520);
  __bf16* EmbTb = (__bf16*)(ws + 131596288);
  __bf16* attnT = Kb;                                  // reuse after S
  float*  logitsN = (float*)Vb;                        // reuse after last S

  dim3 blk(128);

  cvt_f32_bf16<<<512, 256, 0, stream>>>(wq, Wqb, 3 * WSTRIDE);
  cvt_f32_bf16<<<512, 256, 0, stream>>>(wk, Wkb, 3 * WSTRIDE);
  cvt_f32_bf16<<<512, 256, 0, stream>>>(wv, Wvb, 3 * WSTRIDE);
  cvt_f32_bf16<<<512, 256, 0, stream>>>(pp, Pb,  3 * WSTRIDE);
  cvt_f32_bf16<<<64, 256, 0, stream>>>(emb, Embb, 128 * 128);
  transpose_cvt128<<<64, 256, 0, stream>>>(emb, EmbTb);
  cvt_xmidT<<<2048, 256, 0, stream>>>(x, QbT);         // xmidT staged in Q buffer
  copy_edgesT<<<2048, 256, 0, stream>>>(x, h32T, hbT);

  // h cols 64..191:  hT[:, :, 64:192] = (Emb @ x_mid)^T   M=128,N=2048,K=128
  gemm_bf16_lds<<<dim3(32, 2, 16), blk, 0, stream>>>(
      Embb, 0L, 128, QbT, (long)Nn * 128, 128,
      h32T + 64, hbT + 64, HSTRIDE, Dm,
      Nn, 128, 1.0f, F_TRANSC);

  for (int nl = 0; nl < 3; ++nl) {
    const __bf16* wqn = Wqb + nl * WSTRIDE;
    const __bf16* wkn = Wkb + nl * WSTRIDE;
    const __bf16* wvn = Wvb + nl * WSTRIDE;
    const __bf16* pn  = Pb  + nl * WSTRIDE;
    dim3 g(32, 5, 16);  // N=2048/64, M=320/64, batch

    // Q^T = (Wq @ h)^T
    gemm_bf16_lds<<<g, blk, 0, stream>>>(wqn, 0L, Dm, hbT, HSTRIDE, Dm,
        nullptr, QbT, HSTRIDE, Dm, Nn, Dm, 1.0f, F_TRANSC);
    // K = Wk @ h (row-major, column 2047 zeroed: context = first 2047)
    gemm_bf16_lds<<<g, blk, 0, stream>>>(wkn, 0L, Dm, hbT, HSTRIDE, Dm,
        nullptr, Kb, HSTRIDE, Nn, Nn, Dm, 1.0f, F_ZLAST);
    // V = Wv @ h (row-major)
    gemm_bf16_lds<<<g, blk, 0, stream>>>(wvn, 0L, Dm, hbT, HSTRIDE, Dm,
        nullptr, Vb, HSTRIDE, Nn, Nn, Dm, 1.0f, 0);
    // S = (1/2047) * V @ K^T
    gemm_vkt<<<dim3(5, 5, 16), blk, 0, stream>>>(Vb, Kb, Sb, 1.0f / 2047.0f);
    // attn^T = (S @ Q)^T   (reuse K buffer)
    gemm_bf16_lds<<<g, blk, 0, stream>>>(Sb, WSTRIDE, Dm, QbT, HSTRIDE, Dm,
        nullptr, attnT, HSTRIDE, Dm, Nn, Dm, 1.0f, F_TRANSC);
    // h^T += (P @ attn)^T  (f32 accumulate + refresh bf16 mirror)
    gemm_bf16_lds<<<g, blk, 0, stream>>>(pn, 0L, Dm, attnT, HSTRIDE, Dm,
        h32T, hbT, HSTRIDE, Dm, Nn, Dm, 1.0f, F_ACCUM | F_TRANSC);
  }

  // logits[n,c] = (EmbT @ f)^T : A=EmbT(128x128), BT = hT cols 192..319
  gemm_bf16_lds<<<dim3(32, 2, 16), blk, 0, stream>>>(
      EmbTb, 0L, EMBC, hbT + 192, HSTRIDE, Dm,
      logitsN, nullptr, (long)Nn * EMBC, EMBC,
      Nn, EMBC, 1.0f, F_TRANSC);

  softmax_out<<<(B_SZ * Nn) / 256, 256, 0, stream>>>(logitsN, out);
}
